// CrossAttention_53377853554995
// MI455X (gfx1250) — matrix operand order repaired and checked by reading, not compiled, not run on hardware
//
#include <hip/hip_runtime.h>
#include <hip/hip_bf16.h>

// ---------------------------------------------------------------------------
// CDNA5 (gfx1250) bf16 WMMA cross-attention pipeline.
//   B=4, S=2048, D_IN=D_H=D_OUT=512
// ---------------------------------------------------------------------------

typedef __attribute__((ext_vector_type(16))) __bf16 bfx16;
typedef __attribute__((ext_vector_type(8)))  __bf16 bfx8;
typedef __attribute__((ext_vector_type(8)))  float  fx8;

#define WMMA_BF16(A, B, C) \
  __builtin_amdgcn_wmma_f32_16x16x32_bf16(false, (A), false, (B), (short)0, (C), false, false)

// A-matrix fragment (16x32 bf16, MxK), tile row-major with element stride.
// Per ISA 7.12.2: lane L (L<16) row M=L holds K=[0..7] in v[0..3] pairs and
// K=[16..23] in v[4..7]; lane L+16 holds K=[8..15] and K=[24..31].
__device__ __forceinline__ bfx16 load_frag_a(const __bf16* p, int stride) {
  const int lane = (int)(threadIdx.x & 31u);
  const int row = lane & 15, h = lane >> 4;
  const __bf16* base = p + row * stride + h * 8;
  union { bfx16 v; bfx8 q[2]; } u;
  u.q[0] = *(const bfx8*)(base);        // K = h*8 .. h*8+7
  u.q[1] = *(const bfx8*)(base + 16);   // K = 16+h*8 .. 16+h*8+7
  return u.v;
}

// B-matrix fragment (32x16 bf16, KxN), source stored "n-major": element (k,n)
// at p[n*stride + k]. Lanes 0-15 hold K=0..7,16..23 of col n=lane, lanes 16-31 K=8..15,24..31.
__device__ __forceinline__ bfx16 load_frag_b(const __bf16* p, int stride) {
  const int lane = (int)(threadIdx.x & 31u);
  const int col = lane & 15, h = lane >> 4;
  const __bf16* base = p + col * stride + h * 8;
  union { bfx16 v; bfx8 q[2]; } u;
  u.q[0] = *(const bfx8*)(base);        // K = h*8 .. h*8+7
  u.q[1] = *(const bfx8*)(base + 16);   // K = 16+h*8 .. 16+h*8+7
  return u.v;
}

// A fragment built from an f32 tile (probabilities in LDS), converted to bf16.
// Consecutive K pairs are contiguous -> read float2 (ds_load_b64) x8.
__device__ __forceinline__ bfx16 load_frag_a_f32(const float* p, int stride) {
  const int lane = (int)(threadIdx.x & 31u);
  const int row = lane & 15, h = lane >> 4;
  const float* r = p + row * stride + h * 8;
  bfx16 v;
#pragma unroll
  for (int j = 0; j < 4; ++j) {
    float2 lo = *(const float2*)(r + 2 * j);
    float2 hi = *(const float2*)(r + 16 + 2 * j);
    v[2 * j]     = (__bf16)lo.x;
    v[2 * j + 1] = (__bf16)lo.y;
    v[2 * j + 8] = (__bf16)hi.x;
    v[2 * j + 9] = (__bf16)hi.y;
  }
  return v;
}

__device__ __forceinline__ fx8 zero8() {
  fx8 z = {0.f, 0.f, 0.f, 0.f, 0.f, 0.f, 0.f, 0.f};
  return z;
}

// ---------------------------------------------------------------------------
// Shared GEMM core: C[128x64] tile of  A[f32, MxK] @ B[f32, Kx512]  in bf16.
// 256 threads = 8 waves; wave (wid&3, wid>>2) owns a 32x32 sub-tile (2x2 WMMA).
// LDS: A tile row-major stride 40 (padded), B tile n-major stride 40.
// ---------------------------------------------------------------------------
#define LDA_STR 40
#define LDB_STR 40

__device__ __forceinline__ void gemm_tile_core(
    const float* __restrict__ A, const float* __restrict__ Bw, int K,
    int m0, int n0, __bf16* lAs, __bf16* lBs, fx8 acc[2][2]) {
  const int tid = (int)threadIdx.x;
  const int wid = tid >> 5;
  const int wm = (wid & 3) * 32;
  const int wn = (wid >> 2) * 32;

  for (int kb = 0; kb < K; kb += 32) {
    __syncthreads();
    // Stage A: 128 rows x 32 k (f32 -> bf16), float4 coalesced.
#pragma unroll
    for (int r = 0; r < 4; ++r) {
      int slot = tid + r * 256;
      int ar = slot >> 3;
      int ac = (slot & 7) * 4;
      float4 f = *(const float4*)(A + (size_t)(m0 + ar) * K + kb + ac);
      __bf16* d = lAs + ar * LDA_STR + ac;
      d[0] = (__bf16)f.x; d[1] = (__bf16)f.y; d[2] = (__bf16)f.z; d[3] = (__bf16)f.w;
    }
    // Stage B: 32 k x 64 n, transposed into n-major LDS.
#pragma unroll
    for (int r = 0; r < 2; ++r) {
      int slot = tid + r * 256;
      int k = slot >> 4;
      int nc = (slot & 15) * 4;
      float4 f = *(const float4*)(Bw + (size_t)(kb + k) * 512 + n0 + nc);
      lBs[(nc + 0) * LDB_STR + k] = (__bf16)f.x;
      lBs[(nc + 1) * LDB_STR + k] = (__bf16)f.y;
      lBs[(nc + 2) * LDB_STR + k] = (__bf16)f.z;
      lBs[(nc + 3) * LDB_STR + k] = (__bf16)f.w;
    }
    __syncthreads();

    bfx16 a0 = load_frag_a(lAs + wm * LDA_STR, LDA_STR);
    bfx16 a1 = load_frag_a(lAs + (wm + 16) * LDA_STR, LDA_STR);
    bfx16 b0 = load_frag_b(lBs + wn * LDB_STR, LDB_STR);
    bfx16 b1 = load_frag_b(lBs + (wn + 16) * LDB_STR, LDB_STR);
    acc[0][0] = WMMA_BF16(a0, b0, acc[0][0]);
    acc[0][1] = WMMA_BF16(a0, b1, acc[0][1]);
    acc[1][0] = WMMA_BF16(a1, b0, acc[1][0]);
    acc[1][1] = WMMA_BF16(a1, b1, acc[1][1]);
  }
}

// ---------------------------------------------------------------------------
// Kernel 1: 9 fused QKV projections. grid = (N/64, M/128, 9).
//  z = 3*xi + wj : xi selects x1/x2/x3, wj selects (Wq->Q),(Wk->K),(Wv->V^T)
//  Q,K stored bf16 [3][B*S, 512]; V stored transposed bf16 [3][B][512][2048].
// ---------------------------------------------------------------------------
__global__ __launch_bounds__(256) void proj_qkv_kernel(
    const float* __restrict__ x1, const float* __restrict__ x2,
    const float* __restrict__ x3, const float* __restrict__ Wq,
    const float* __restrict__ bq, const float* __restrict__ Wk,
    const float* __restrict__ bk, const float* __restrict__ Wv,
    const float* __restrict__ bv, __bf16* __restrict__ Qo,
    __bf16* __restrict__ Ko, __bf16* __restrict__ Vto) {
  __shared__ __attribute__((aligned(16))) __bf16 lAs[128 * LDA_STR];
  __shared__ __attribute__((aligned(16))) __bf16 lBs[64 * LDB_STR];

  const int z = (int)blockIdx.z;
  const int xi = z / 3, wj = z % 3;
  const float* A    = (xi == 0) ? x1 : ((xi == 1) ? x2 : x3);
  const float* W    = (wj == 0) ? Wq : ((wj == 1) ? Wk : Wv);
  const float* bias = (wj == 0) ? bq : ((wj == 1) ? bk : bv);
  __bf16* out = ((wj == 0) ? Qo : ((wj == 1) ? Ko : Vto)) + (size_t)xi * (8192ull * 512ull);
  const bool tr = (wj == 2);

  const int m0 = (int)blockIdx.y * 128;
  const int n0 = (int)blockIdx.x * 64;

  fx8 acc[2][2] = {{zero8(), zero8()}, {zero8(), zero8()}};
  gemm_tile_core(A, W, 512, m0, n0, lAs, lBs, acc);

  const int tid = (int)threadIdx.x;
  const int lane = tid & 31, hlf = lane >> 4, ln = lane & 15;
  const int wid = tid >> 5, wm = (wid & 3) * 32, wn = (wid >> 2) * 32;
#pragma unroll
  for (int mt = 0; mt < 2; ++mt)
#pragma unroll
    for (int nt = 0; nt < 2; ++nt)
#pragma unroll
      for (int i = 0; i < 8; ++i) {
        int row = m0 + wm + mt * 16 + hlf * 8 + i;
        int col = n0 + wn + nt * 16 + ln;
        float v = acc[mt][nt][i] + bias[col];
        if (!tr) {
          out[(size_t)row * 512 + col] = (__bf16)v;
        } else {
          int b = row >> 11, s = row & 2047;  // S = 2048
          out[((size_t)(b * 512 + col) << 11) + s] = (__bf16)v;
        }
      }
}

// ---------------------------------------------------------------------------
// Kernel 2: flash attention, Out[b, q, qoff+ :512] (+)= softmax(QK^T/√d) V.
// grid = (S/32, B); 256 threads = 8 waves. Wave w: scores for 16 keys of the
// 128-key block, then P·V for output columns [w*64, w*64+64).
// Q tile is staged with GLOBAL_LOAD_ASYNC_TO_LDS_B128 (ASYNCcnt path).
// ---------------------------------------------------------------------------
#define QSTR 520   // 512 + 8 pad (keeps 16B alignment: 520*2 = 1040 = 16*65)
#define SSTR 136   // 128 + 8 pad

__global__ __launch_bounds__(256) void attn_kernel(
    const __bf16* __restrict__ Qg, const __bf16* __restrict__ Kg,
    const __bf16* __restrict__ Vtg, float* __restrict__ fused,
    int qoff, int accumulate) {
  __shared__ __attribute__((aligned(16))) __bf16 q_s[32 * QSTR];
  __shared__ __attribute__((aligned(16))) float s_tile[32 * SSTR];
  __shared__ float row_m[32], row_l[32], row_scale[32];

  const int tid = (int)threadIdx.x;
  const int bb = (int)blockIdx.y;
  const int q0 = (int)blockIdx.x * 32;
  const int lane = tid & 31, hlf = lane >> 4, ln = lane & 15;
  const int w = tid >> 5;
  const float SCALE = 0.044194173824159216f;  // 1/sqrt(512)

  // Stage Q tile: 32 rows x 512 bf16 via async global->LDS b128 copies.
  // 2048 16-byte chunks, 8 per thread; both sides 16B aligned.
  {
    const unsigned int lds_base = (unsigned int)(uintptr_t)q_s;
#pragma unroll
    for (int r8 = 0; r8 < 8; ++r8) {
      int idx = tid + r8 * 256;
      int r = idx >> 6;           // row 0..31
      int c = (idx & 63) * 8;     // element offset, 8 bf16 = 16 B
      unsigned int ldsa = lds_base + (unsigned int)(r * QSTR + c) * 2u;
      unsigned long long ga =
          (unsigned long long)(uintptr_t)(Qg + ((size_t)bb * 2048 + q0 + r) * 512 + c);
      asm volatile("global_load_async_to_lds_b128 %0, %1, off"
                   :: "v"(ldsa), "v"(ga)
                   : "memory");
    }
    asm volatile("s_wait_asynccnt 0" ::: "memory");
  }
  if (tid < 32) {
    row_m[tid] = -__builtin_inff();
    row_l[tid] = 0.0f;
  }
  fx8 acc[2][4] = {{zero8(), zero8(), zero8(), zero8()},
                   {zero8(), zero8(), zero8(), zero8()}};
  __syncthreads();

  for (int kb = 0; kb < 2048; kb += 128) {
    // --- scores: S_w[32 x 16] = Q[32,512] @ K[keys]^T, B-frags from global ---
    fx8 s0 = zero8(), s1 = zero8();
    const __bf16* kbase = Kg + ((size_t)bb * 2048 + kb + w * 16) * 512;
#pragma unroll
    for (int kk = 0; kk < 512; kk += 32) {
      bfx16 aq0 = load_frag_a(q_s + kk, QSTR);
      bfx16 aq1 = load_frag_a(q_s + 16 * QSTR + kk, QSTR);
      bfx16 bkf = load_frag_b(kbase + kk, 512);
      s0 = WMMA_BF16(aq0, bkf, s0);
      s1 = WMMA_BF16(aq1, bkf, s1);
    }

    __syncthreads();  // previous iteration's s_tile consumers are done
#pragma unroll
    for (int i = 0; i < 8; ++i) {
      s_tile[(hlf * 8 + i) * SSTR + w * 16 + ln] = s0[i] * SCALE;
      s_tile[(16 + hlf * 8 + i) * SSTR + w * 16 + ln] = s1[i] * SCALE;
    }
    __syncthreads();

    // --- online softmax bookkeeping: thread t owns query row t ---
    if (tid < 32) {
      float* sr = s_tile + tid * SSTR;
      float mo = row_m[tid], lo = row_l[tid];
      float bm = -__builtin_inff();
      for (int k2 = 0; k2 < 128; ++k2) bm = fmaxf(bm, sr[k2]);
      float mn = fmaxf(mo, bm);
      float sc = __expf(mo - mn);  // exp(-inf) = 0 on first block
      float sum = 0.0f;
      for (int k2 = 0; k2 < 128; ++k2) {
        float p = __expf(sr[k2] - mn);
        sr[k2] = p;
        sum += p;
      }
      row_m[tid] = mn;
      row_l[tid] = lo * sc + sum;
      row_scale[tid] = sc;
    }
    __syncthreads();

    // --- rescale accumulators by exp(m_old - m_new) ---
#pragma unroll
    for (int i = 0; i < 8; ++i) {
      float sA = row_scale[hlf * 8 + i];
      float sB = row_scale[16 + hlf * 8 + i];
#pragma unroll
      for (int nt = 0; nt < 4; ++nt) {
        acc[0][nt][i] *= sA;
        acc[1][nt][i] *= sB;
      }
    }

    // --- P·V: A = probs (LDS f32 -> bf16), B = V^T stream from global ---
#pragma unroll
    for (int c = 0; c < 4; ++c) {
      bfx16 p0 = load_frag_a_f32(s_tile + c * 32, SSTR);
      bfx16 p1 = load_frag_a_f32(s_tile + 16 * SSTR + c * 32, SSTR);
#pragma unroll
      for (int nt = 0; nt < 4; ++nt) {
        const __bf16* vb =
            Vtg + ((size_t)bb * 512 + w * 64 + nt * 16) * 2048 + kb + c * 32;
        bfx16 bvf = load_frag_b(vb, 2048);
        acc[0][nt] = WMMA_BF16(p0, bvf, acc[0][nt]);
        acc[1][nt] = WMMA_BF16(p1, bvf, acc[1][nt]);
      }
    }
    // Prefetch next key block into caches (global_prefetch_b8).
    if (kb + 128 < 2048)
      __builtin_prefetch(Kg + ((size_t)bb * 2048 + kb + 128 + w * 16) * 512, 0, 3);
  }
  __syncthreads();

  // --- epilogue: divide by l, add into fused[b, q, qoff + col] ---
#pragma unroll
  for (int mt = 0; mt < 2; ++mt)
#pragma unroll
    for (int i = 0; i < 8; ++i) {
      int row = mt * 16 + hlf * 8 + i;
      float invl = 1.0f / row_l[row];
#pragma unroll
      for (int nt = 0; nt < 4; ++nt) {
        int col = w * 64 + nt * 16 + ln;
        float v = acc[mt][nt][i] * invl;
        float* o = fused + ((size_t)bb * 2048 + q0 + row) * 1536 + qoff + col;
        if (accumulate) v += *o;
        *o = v;
      }
    }
}

// ---------------------------------------------------------------------------
// Kernel 3: output projection  d_out[8192,512] = fused[8192,1536] @ Wo + bo.
// ---------------------------------------------------------------------------
__global__ __launch_bounds__(256) void proj_out_kernel(
    const float* __restrict__ fused, const float* __restrict__ Wo,
    const float* __restrict__ bo, float* __restrict__ out) {
  __shared__ __attribute__((aligned(16))) __bf16 lAs[128 * LDA_STR];
  __shared__ __attribute__((aligned(16))) __bf16 lBs[64 * LDB_STR];

  const int m0 = (int)blockIdx.y * 128;
  const int n0 = (int)blockIdx.x * 64;

  fx8 acc[2][2] = {{zero8(), zero8()}, {zero8(), zero8()}};
  gemm_tile_core(fused, Wo, 1536, m0, n0, lAs, lBs, acc);

  const int tid = (int)threadIdx.x;
  const int lane = tid & 31, hlf = lane >> 4, ln = lane & 15;
  const int wid = tid >> 5, wm = (wid & 3) * 32, wn = (wid >> 2) * 32;
#pragma unroll
  for (int mt = 0; mt < 2; ++mt)
#pragma unroll
    for (int nt = 0; nt < 2; ++nt)
#pragma unroll
      for (int i = 0; i < 8; ++i) {
        int row = m0 + wm + mt * 16 + hlf * 8 + i;
        int col = n0 + wn + nt * 16 + ln;
        out[(size_t)row * 512 + col] = acc[mt][nt][i] + bo[col];
      }
}

// ---------------------------------------------------------------------------
// Host launch.
// ---------------------------------------------------------------------------
extern "C" void kernel_launch(void* const* d_in, const int* in_sizes, int n_in,
                              void* d_out, int out_size, void* d_ws,
                              size_t ws_size, hipStream_t stream) {
  const float* x1 = (const float*)d_in[0];
  const float* x2 = (const float*)d_in[1];
  const float* x3 = (const float*)d_in[2];
  const float* Wq = (const float*)d_in[3];
  const float* bq = (const float*)d_in[4];
  const float* Wk = (const float*)d_in[5];
  const float* bk = (const float*)d_in[6];
  const float* Wv = (const float*)d_in[7];
  const float* bv = (const float*)d_in[8];
  const float* Wo = (const float*)d_in[9];
  const float* bo = (const float*)d_in[10];

  const size_t QKV = 8192ull * 512ull;  // elements per projected stream
  __bf16* Qw = (__bf16*)d_ws;           // [3][8192][512]   bf16
  __bf16* Kw = Qw + 3 * QKV;            // [3][8192][512]   bf16
  __bf16* Vw = Kw + 3 * QKV;            // [3][B][512][2048] bf16 (transposed)
  float* fused = (float*)(Vw + 3 * QKV);  // [8192][1536]   f32

  // 1) Nine QKV projections in one launch (grid.z selects GEMM).
  proj_qkv_kernel<<<dim3(8, 64, 9), 256, 0, stream>>>(
      x1, x2, x3, Wq, bq, Wk, bk, Wv, bv, Qw, Kw, Vw);

  // 2) Six cross attentions; first pass per q-stream writes, second adds.
  dim3 ag(64, 4);
  attn_kernel<<<ag, 256, 0, stream>>>(Qw + 0 * QKV, Kw + 1 * QKV, Vw + 1 * QKV, fused, 0, 0);
  attn_kernel<<<ag, 256, 0, stream>>>(Qw + 0 * QKV, Kw + 2 * QKV, Vw + 2 * QKV, fused, 0, 1);
  attn_kernel<<<ag, 256, 0, stream>>>(Qw + 1 * QKV, Kw + 0 * QKV, Vw + 0 * QKV, fused, 512, 0);
  attn_kernel<<<ag, 256, 0, stream>>>(Qw + 1 * QKV, Kw + 2 * QKV, Vw + 2 * QKV, fused, 512, 1);
  attn_kernel<<<ag, 256, 0, stream>>>(Qw + 2 * QKV, Kw + 0 * QKV, Vw + 0 * QKV, fused, 1024, 0);
  attn_kernel<<<ag, 256, 0, stream>>>(Qw + 2 * QKV, Kw + 1 * QKV, Vw + 1 * QKV, fused, 1024, 1);

  // 3) Output projection.
  proj_out_kernel<<<dim3(8, 64), 256, 0, stream>>>(fused, Wo, bo, (float*)d_out);
}
